// Generator_55791625175543
// MI455X (gfx1250) — matrix operand order repaired and checked by reading, not compiled, not run on hardware
//
#include <hip/hip_runtime.h>

// MI455X / gfx1250, wave32. Fused point-upsampling MLP (f16 WMMA, f32 accum,
// all activations LDS-resident, weights pre-swizzled into per-lane fragment
// order) + register-resident farthest-point sampling.

typedef __attribute__((ext_vector_type(16))) _Float16 v16h;
typedef __attribute__((ext_vector_type(8)))  _Float16 v8h;
typedef __attribute__((ext_vector_type(8)))  float    v8f;

#define BATCH 16
#define NPTS  2048
#define RUP   5
#define MPTS  (NPTS * RUP)   // 10240 upsampled points per batch
#define NSAMP 8192           // FPS output count
#define TILE  64             // input points per workgroup (4 waves x 16 rows)

// ---------------------------------------------------------------------------
// WMMA fragment helpers (layouts per CDNA5 ISA sec 7.12.2)
// ---------------------------------------------------------------------------

// A-matrix 16x32 f16 from row-major LDS (row stride `stride` halfs).
// lanes 0-15: row=lane,     halfs 0..7 -> K 0..7,  halfs 8..15 -> K 16..23
// lanes 16-31: row=lane-16, halfs 0..7 -> K 8..15, halfs 8..15 -> K 24..31
// Both 8-half groups are 16-byte aligned -> two ds_load_b128 per lane.
__device__ __forceinline__ v16h load_a_frag(const _Float16* A, int stride) {
    int lane = threadIdx.x & 31;
    const _Float16* p = A + (lane & 15) * stride + ((lane < 16) ? 0 : 8);
    v8h lo = *(const v8h*)(p);
    v8h hi = *(const v8h*)(p + 16);
    v16h a;
#pragma unroll
    for (int i = 0; i < 8; ++i) { a[i] = lo[i]; a[8 + i] = hi[i]; }
    return a;
}

// B-matrix 32x16 f16 from LDS pre-swizzled into per-lane fragment order:
// tile base -> 32 lanes x 16 contiguous halfs (32B per lane, conflict-free).
__device__ __forceinline__ v16h load_b_frag_sw(const _Float16* Bsw) {
    int lane = threadIdx.x & 31;
    const _Float16* p = Bsw + lane * 16;
    v8h lo = *(const v8h*)(p);
    v8h hi = *(const v8h*)(p + 8);
    v16h b;
#pragma unroll
    for (int i = 0; i < 8; ++i) { b[i] = lo[i]; b[8 + i] = hi[i]; }
    return b;
}

// C/D f32 16x16: VGPR v -> row v (lanes 0-15) / row v+8 (lanes 16-31), col=lane%16
template <bool RELU>
__device__ __forceinline__ void store_d_frag(_Float16* D, int stride, v8f acc) {
    int lane = threadIdx.x & 31;
    int col  = lane & 15;
    int rb   = (lane < 16) ? 0 : 8;
#pragma unroll
    for (int v = 0; v < 8; ++v) {
        float f = acc[v];
        if (RELU) f = f > 0.f ? f : 0.f;
        D[(rb + v) * stride + col] = (_Float16)f;
    }
}

// Gather a row-major f32 weight matrix W[K x N] from global into LDS,
// converted to f16 and swizzled to B-fragment order: 512 halfs per 32x16
// (KxN) tile; within a tile, lane L owns 16 contiguous halfs:
//   lane<16:  col=L,    K = k0+0..7,16..23 ;  lane>=16: col=L-16, K = k0+8..15,24..31
__device__ __forceinline__ void preload_b_swizzled(const float* __restrict__ W,
                                                   _Float16* dst, int K, int N,
                                                   int tid, int nthreads) {
    const int ntn = N >> 4;                 // N tiles of 16 columns
    const int total = K * N;
    for (int d = tid; d < total; d += nthreads) {
        int tile   = d >> 9;                // 512 halfs per tile
        int within = d & 511;
        int lane   = within >> 4;
        int j      = within & 15;
        int kt     = tile / ntn;
        int ct     = tile - kt * ntn;
        int k      = (kt << 5) + ((lane & 16) ? 8 : 0) + (j & 7) + ((j & 8) ? 16 : 0);
        int n      = (ct << 4) + (lane & 15);
        dst[d] = (_Float16)W[k * N + n];
    }
}

// ---------------------------------------------------------------------------
// Kernel 1: fused MLP 3->64->128 ->(x5 replicas: 128->128->64->3) -> coords
// One workgroup = 128 threads (4 waves) handles a 64-point tile of one batch.
// ---------------------------------------------------------------------------
__global__ __launch_bounds__(128) void mlp_coords_kernel(
    const float* __restrict__ points,                       // [B,N,3]
    const float* __restrict__ w1,  const float* __restrict__ b1,    // [3,64]
    const float* __restrict__ w2,  const float* __restrict__ b2,    // [64,128]
    const float* __restrict__ wup, const float* __restrict__ bup,   // [129,128]
    const float* __restrict__ wc1, const float* __restrict__ bc1,   // [128,64]
    const float* __restrict__ wc2, const float* __restrict__ bc2,   // [64,3]
    float* __restrict__ coords)                             // [B, M, 3]
{
    __shared__ __align__(16) _Float16 sW2 [64 * 128];    // swizzled (2 K-tiles x 8 N-tiles)
    __shared__ __align__(16) _Float16 sWup[128 * 128];   // swizzled (4 x 8), rows 0..127
    __shared__ __align__(16) _Float16 sWc1[128 * 64];    // swizzled (4 x 4)
    __shared__ _Float16 sWupLast[128];                   // w_up row 128 (code channel)
    __shared__ float    sW1[3 * 64];
    __shared__ float    sB1[64], sB2[128], sBup[128], sBc1[64];
    __shared__ float    sWc2[64 * 3], sBc2[4];
    __shared__ float    sPts[TILE * 3];
    __shared__ __align__(16) _Float16 sF1[TILE * 64];
    __shared__ __align__(16) _Float16 sF2[TILE * 128];
    __shared__ __align__(16) _Float16 sH [TILE * 128];
    __shared__ __align__(16) _Float16 sC [TILE * 64];

    const int tid  = threadIdx.x;
    const int lane = tid & 31;
    const int wv   = tid >> 5;                 // wave id 0..3 -> 16-row strip
    const int b    = blockIdx.x / (NPTS / TILE);
    const int n0   = (blockIdx.x % (NPTS / TILE)) * TILE;

    // Cooperative weight preload: f32 -> f16 + B-fragment swizzle (L2-cached)
    preload_b_swizzled(w2,  sW2,  64,  128, tid, 128);
    preload_b_swizzled(wup, sWup, 128, 128, tid, 128);
    preload_b_swizzled(wc1, sWc1, 128, 64,  tid, 128);
    for (int i = tid; i < 128;       i += 128) sWupLast[i] = (_Float16)wup[128 * 128 + i];
    for (int i = tid; i < 3 * 64;    i += 128) sW1[i]  = w1[i];
    for (int i = tid; i < 64;        i += 128) sB1[i]  = b1[i];
    for (int i = tid; i < 128;       i += 128) sB2[i]  = b2[i];
    for (int i = tid; i < 128;       i += 128) sBup[i] = bup[i];
    for (int i = tid; i < 64;        i += 128) sBc1[i] = bc1[i];
    for (int i = tid; i < 64 * 3;    i += 128) sWc2[i] = wc2[i];
    for (int i = tid; i < 3;         i += 128) sBc2[i] = bc2[i];
    for (int i = tid; i < TILE * 3;  i += 128) sPts[i] = points[((size_t)b * NPTS + n0) * 3 + i];
    __syncthreads();

    // ---- L1 (K=3, VALU): f1 = relu(points @ w1 + b1) -------------------
    for (int i = tid; i < TILE * 64; i += 128) {
        int row = i >> 6, col = i & 63;
        float x = sPts[row * 3 + 0], y = sPts[row * 3 + 1], z = sPts[row * 3 + 2];
        float v = x * sW1[col] + y * sW1[64 + col] + z * sW1[128 + col] + sB1[col];
        sF1[row * 64 + col] = (_Float16)(v > 0.f ? v : 0.f);
    }
    __syncthreads();

    // ---- L2 (WMMA, K=64): f2 = relu(f1 @ w2 + b2) ----------------------
    {
        const _Float16* A = sF1 + wv * 16 * 64;
#pragma unroll
        for (int ct = 0; ct < 8; ++ct) {
            float bias = sB2[ct * 16 + (lane & 15)];
            v8f acc;
#pragma unroll
            for (int i = 0; i < 8; ++i) acc[i] = bias;
#pragma unroll
            for (int kt = 0; kt < 2; ++kt) {
                v16h a  = load_a_frag(A + kt * 32, 64);
                v16h bm = load_b_frag_sw(sW2 + ((kt * 8) + ct) * 512);
                acc = __builtin_amdgcn_wmma_f32_16x16x32_f16(false, a, false, bm,
                                                             (short)0, acc, false, false);
            }
            store_d_frag<true>(sF2 + wv * 16 * 128 + ct * 16, 128, acc);
        }
    }
    __syncthreads();

    // ---- Replica loop: code channel folded into bias (rank-1 trick) ----
    for (int rep = 0; rep < RUP; ++rep) {
        const float code = (float)rep / (float)RUP;

        // h = relu(f2 @ wup[0:128] + (bup + code*wup[128]))   (WMMA, K=128)
        {
            const _Float16* A = sF2 + wv * 16 * 128;
#pragma unroll
            for (int ct = 0; ct < 8; ++ct) {
                int col = ct * 16 + (lane & 15);
                float bias = sBup[col] + code * (float)sWupLast[col];
                v8f acc;
#pragma unroll
                for (int i = 0; i < 8; ++i) acc[i] = bias;
#pragma unroll
                for (int kt = 0; kt < 4; ++kt) {
                    v16h a  = load_a_frag(A + kt * 32, 128);
                    v16h bm = load_b_frag_sw(sWup + ((kt * 8) + ct) * 512);
                    acc = __builtin_amdgcn_wmma_f32_16x16x32_f16(false, a, false, bm,
                                                                 (short)0, acc, false, false);
                }
                store_d_frag<true>(sH + wv * 16 * 128 + ct * 16, 128, acc);
            }
        }
        __syncthreads();

        // c = relu(h @ wc1 + bc1)   (WMMA, K=128)
        {
            const _Float16* A = sH + wv * 16 * 128;
#pragma unroll
            for (int ct = 0; ct < 4; ++ct) {
                float bias = sBc1[ct * 16 + (lane & 15)];
                v8f acc;
#pragma unroll
                for (int i = 0; i < 8; ++i) acc[i] = bias;
#pragma unroll
                for (int kt = 0; kt < 4; ++kt) {
                    v16h a  = load_a_frag(A + kt * 32, 128);
                    v16h bm = load_b_frag_sw(sWc1 + ((kt * 4) + ct) * 512);
                    acc = __builtin_amdgcn_wmma_f32_16x16x32_f16(false, a, false, bm,
                                                                 (short)0, acc, false, false);
                }
                store_d_frag<true>(sC + wv * 16 * 64 + ct * 16, 64, acc);
            }
        }
        __syncthreads();

        // coords = c @ wc2 + bc2   (N=3, VALU) -> global
        for (int i = tid; i < TILE * 3; i += 128) {
            int row = i / 3, col = i % 3;
            float acc = sBc2[col];
#pragma unroll 8
            for (int k = 0; k < 64; ++k)
                acc += (float)sC[row * 64 + k] * sWc2[k * 3 + col];
            coords[((size_t)b * MPTS + rep * NPTS + n0 + row) * 3 + col] = acc;
        }
        __syncthreads();   // before sH / sC are overwritten next replica
    }
}

// ---------------------------------------------------------------------------
// Kernel 2: farthest point sampling (8192 serial steps) + fused gather.
// One 1024-thread block per batch; each thread keeps its 10 points and the
// running min-distance entirely in registers. wave32 shfl argmax + LDS tree.
// ---------------------------------------------------------------------------
#define FPS_T 1024
#define CHUNK (MPTS / FPS_T)   // 10

__global__ __launch_bounds__(FPS_T) void fps_kernel(
    const float* __restrict__ coords,   // [B, M, 3]
    float* __restrict__ out)            // [B, NSAMP, 3]
{
    const int b   = blockIdx.x;
    const int tid = threadIdx.x;
    const int lane = tid & 31;
    const float* cb = coords + (size_t)b * MPTS * 3;

    float px[CHUNK], py[CHUNK], pz[CHUNK], dist[CHUNK];
    const int base = tid * CHUNK;
#pragma unroll
    for (int i = 0; i < CHUNK; ++i) {
        px[i] = cb[(base + i) * 3 + 0];
        py[i] = cb[(base + i) * 3 + 1];
        pz[i] = cb[(base + i) * 3 + 2];
        dist[i] = 1e10f;
    }

    __shared__ float sval[32];
    __shared__ int   sidx[32];
    __shared__ float scent[3];
    __shared__ int   sfar;

    int far = 0;   // reference seeds at index 0
    for (int s = 0; s < NSAMP; ++s) {
        // Owner of current pick publishes its coordinates
        if (far >= base && far < base + CHUNK) {
            int l = far - base;
            scent[0] = px[l]; scent[1] = py[l]; scent[2] = pz[l];
        }
        __syncthreads();
        float cx = scent[0], cy = scent[1], cz = scent[2];
        if (tid == 0) {   // fused gather: out[b][s] = coords[b][far]
            out[((size_t)b * NSAMP + s) * 3 + 0] = cx;
            out[((size_t)b * NSAMP + s) * 3 + 1] = cy;
            out[((size_t)b * NSAMP + s) * 3 + 2] = cz;
        }

        // Update register-resident min distances; local first-max argmax
        float best = -1.f; int bi = 0;
#pragma unroll
        for (int i = 0; i < CHUNK; ++i) {
            float dx = px[i] - cx, dy = py[i] - cy, dz = pz[i] - cz;
            float d  = dx * dx + dy * dy + dz * dz;
            float nd = fminf(dist[i], d);
            dist[i]  = nd;
            if (nd > best) { best = nd; bi = base + i; }   // ascending -> first max
        }

        // wave32 argmax (prefer lower index on ties, matching jnp.argmax)
#pragma unroll
        for (int off = 16; off > 0; off >>= 1) {
            float ov = __shfl_xor(best, off);
            int   oi = __shfl_xor(bi,   off);
            if (ov > best || (ov == best && oi < bi)) { best = ov; bi = oi; }
        }
        if (lane == 0) { sval[tid >> 5] = best; sidx[tid >> 5] = bi; }
        __syncthreads();
        if (tid < 32) {
            best = sval[tid]; bi = sidx[tid];
#pragma unroll
            for (int off = 16; off > 0; off >>= 1) {
                float ov = __shfl_xor(best, off);
                int   oi = __shfl_xor(bi,   off);
                if (ov > best || (ov == best && oi < bi)) { best = ov; bi = oi; }
            }
            if (tid == 0) sfar = bi;
        }
        __syncthreads();
        far = sfar;
    }
}

// ---------------------------------------------------------------------------
extern "C" void kernel_launch(void* const* d_in, const int* in_sizes, int n_in,
                              void* d_out, int out_size, void* d_ws, size_t ws_size,
                              hipStream_t stream) {
    (void)in_sizes; (void)n_in; (void)out_size; (void)ws_size;
    const float* points = (const float*)d_in[0];
    const float* w1   = (const float*)d_in[1];
    const float* b1   = (const float*)d_in[2];
    const float* w2   = (const float*)d_in[3];
    const float* b2   = (const float*)d_in[4];
    const float* wup  = (const float*)d_in[5];
    const float* bup  = (const float*)d_in[6];
    const float* wc1  = (const float*)d_in[7];
    const float* bc1  = (const float*)d_in[8];
    const float* wc2  = (const float*)d_in[9];
    const float* bc2  = (const float*)d_in[10];
    // d_in[11] = up_ratio_real (5), d_in[12] = out_num_point (8192): hardcoded.

    float* coords = (float*)d_ws;      // [B, 10240, 3] f32 = ~1.97 MB scratch
    float* out    = (float*)d_out;     // [B, 8192, 3]  f32

    mlp_coords_kernel<<<dim3(BATCH * (NPTS / TILE)), dim3(128), 0, stream>>>(
        points, w1, b1, w2, b2, wup, bup, wc1, bc1, wc2, bc2, coords);
    fps_kernel<<<dim3(BATCH), dim3(FPS_T), 0, stream>>>(coords, out);
}
